// RNN_57707180589180
// MI455X (gfx1250) — compile-verified, matrix operand order checked by reading, not run on hardware
//
#include <hip/hip_runtime.h>

#define SEQ   2048
#define BATCH 256
#define INDIM 64
#define HID   128

typedef __attribute__((ext_vector_type(2)))  float          v2f;
typedef __attribute__((ext_vector_type(8)))  float          v8f;
typedef __attribute__((ext_vector_type(16))) __bf16         v16bf;
typedef __attribute__((ext_vector_type(16))) unsigned short v16us;
typedef __attribute__((ext_vector_type(8)))  unsigned short v8us;

__device__ __forceinline__ unsigned short f32_to_bf16(float f) {
  unsigned int u = __builtin_bit_cast(unsigned int, f);
  u += 0x7FFFu + ((u >> 16) & 1u);           // round-to-nearest-even
  return (unsigned short)(u >> 16);
}

__device__ __forceinline__ float fast_tanh(float x) {
  // tanh(x) = 1 - 2/(exp2(2*log2(e)*x)+1); v_exp_f32 + v_rcp_f32.
  float e = __builtin_amdgcn_exp2f(x * 2.885390081777927f); // 2*log2(e)
  float r = __builtin_amdgcn_rcpf(e + 1.0f);
  return 1.0f - 2.0f * r;
}

// ---------------------------------------------------------------------------
// Kernel 1: xp[s,b,h] = sum_k x[s,b,k]*Wx[h,k] + bx[h] + bh[h]
// f32 WMMA 16x16x4. One block = one 16-row M tile; 8 waves = 8 N tiles of 16.
// Output written into d_out's h_seq region (consumed in place by the scan).
// ---------------------------------------------------------------------------
__global__ __launch_bounds__(256) void rnn_proj_kernel(
    const float* __restrict__ x, const float* __restrict__ Wx,
    const float* __restrict__ bx, const float* __restrict__ bh,
    float* __restrict__ out)
{
  const int  lane = threadIdx.x & 31;
  const int  wv   = threadIdx.x >> 5;
  const int  hf   = lane >> 4;       // 0: lanes 0-15, 1: lanes 16-31
  const int  lc   = lane & 15;
  const long row0 = (long)blockIdx.x * 16;
  const int  n    = wv * 16 + lc;    // output column

  // B = Wx^T tiles: 16 K-chunks of 4.  B[k][n] = Wx[n][k].
  // f32 B layout (v2f/lane): lanes 0-15 hold K={kb,kb+1}, lanes 16-31 K={kb+2,kb+3}.
  v2f Bc[16];
  const float* wrow = Wx + (long)n * INDIM;
#pragma unroll
  for (int c = 0; c < 16; ++c)
    Bc[c] = *(const v2f*)(wrow + 4 * c + 2 * hf);

  const float bias = bx[n] + bh[n];

  // A layout (v2f/lane): lane m holds {K=kb,kb+1}, lane 16+m holds {kb+2,kb+3}.
  const float* xrow = x + (row0 + lc) * INDIM;
  v8f acc = {};
#pragma unroll
  for (int c = 0; c < 16; ++c) {
    v2f Ac = *(const v2f*)(xrow + 4 * c + 2 * hf);
    acc = __builtin_amdgcn_wmma_f32_16x16x4_f32(false, Ac, false, Bc[c],
                                                (short)0, acc, false, false);
  }

  // D layout: VGPR r -> row r (+8 for lanes 16-31), col = lane%16.
  float* orow = out + row0 * HID + n;
#pragma unroll
  for (int r = 0; r < 8; ++r)
    orow[(long)(r + 8 * hf) * HID] = acc[r] + bias;
}

// ---------------------------------------------------------------------------
// Kernel 2: sequential scan  h = tanh(h @ Wh^T + xp[s])
// 16 blocks (batch tiles of 16) x 8 waves (hidden tiles of 16).
// bf16 WMMA 16x16x32, K=128 -> 4 WMMA per wave per step.
// h double-buffered in LDS (bf16); one barrier per step.
// xp is software-pipelined one step ahead so its global-load latency is
// covered by a full step of WMMA/LDS/barrier work.
// ---------------------------------------------------------------------------
__global__ __launch_bounds__(256) void rnn_scan_kernel(
    const float* __restrict__ Wh, float* __restrict__ out)
{
  __shared__ __align__(16) unsigned short hbuf[2][16 * HID];

  const int lane = threadIdx.x & 31;
  const int wv   = threadIdx.x >> 5;
  const int hf   = lane >> 4;
  const int lc   = lane & 15;
  const int n0   = wv * 16;
  const int nn   = n0 + lc;          // this lane's output column
  const int b0   = blockIdx.x * 16;  // batch tile

  // h0 = 0
  for (int i = threadIdx.x; i < 16 * HID; i += 256) {
    hbuf[0][i] = 0;
    hbuf[1][i] = 0;
  }

  // Preload B = Wh^T in bf16 WMMA-B layout, 4 chunks of K=32.
  // B[k][n] = Wh[n][k]; per lane: N = nn, K in {kb..kb+7} u {kb+16..kb+23},
  // kb = 32c + 8*hf  (elements 0-7 -> first run, 8-15 -> second run).
  v16bf Bw[4];
  const float* whrow = Wh + (long)nn * HID;
#pragma unroll
  for (int c = 0; c < 4; ++c) {
    const int kb = 32 * c + 8 * hf;
    v16us t;
#pragma unroll
    for (int e = 0; e < 8; ++e) t[e]     = f32_to_bf16(whrow[kb + e]);
#pragma unroll
    for (int e = 0; e < 8; ++e) t[8 + e] = f32_to_bf16(whrow[kb + 16 + e]);
    Bw[c] = __builtin_bit_cast(v16bf, t);
  }

  // Per-lane base pointer for the D-layout xp/h slice of this workgroup.
  float* base = out + (long)b0 * HID + nn;
  const long srowstride = (long)BATCH * HID;

  __syncthreads();

  // Prologue: prefetch xp for s = 0.
  float xp[8];
#pragma unroll
  for (int r = 0; r < 8; ++r)
    xp[r] = base[(long)(r + 8 * hf) * HID];

  int p = 0;
  for (int s = 0; s < SEQ; ++s) {
    // Prefetch xp for step s+1 (independent of this step's dependency chain;
    // clamped at the tail -- the extra read is lane-private and unused).
    const int  sn    = (s + 1 < SEQ) ? (s + 1) : s;
    float*     xcur  = base + (long)s  * srowstride;
    float*     xnext = base + (long)sn * srowstride;
    float xpn[8];
#pragma unroll
    for (int r = 0; r < 8; ++r)
      xpn[r] = xnext[(long)(r + 8 * hf) * HID];

    // A = h tile from LDS buffer p, bf16 A layout:
    // lane m (m=lc) holds M=m, K in {ka..ka+7} u {ka+16..ka+23}, ka = 32c + 8*hf.
    v8f acc = {};
    const unsigned short* hb = hbuf[p];
#pragma unroll
    for (int c = 0; c < 4; ++c) {
      const int ka = 32 * c + 8 * hf;
      const unsigned short* arow = hb + lc * HID + ka;
      v8us lo = *(const v8us*)(arow);        // ds_load_b128
      v8us hi = *(const v8us*)(arow + 16);   // ds_load_b128
      v16us a16 = __builtin_shufflevector(lo, hi, 0, 1, 2, 3, 4, 5, 6, 7,
                                          8, 9, 10, 11, 12, 13, 14, 15);
      v16bf A = __builtin_bit_cast(v16bf, a16);
      acc = __builtin_amdgcn_wmma_f32_16x16x32_bf16(false, A, false, Bw[c],
                                                    (short)0, acc, false, false);
    }

    // h = tanh(acc + xp): write f32 to h_seq[s] (in place over xp),
    // write bf16 to the other LDS buffer for the next step.
    unsigned short* wb = hbuf[p ^ 1];
#pragma unroll
    for (int r = 0; r < 8; ++r) {
      const int mr = r + 8 * hf;
      float t = fast_tanh(acc[r] + xp[r]);
      xcur[(long)mr * HID] = t;
      wb[mr * HID + nn] = f32_to_bf16(t);
      if (s == SEQ - 1) {
        out[(long)SEQ * BATCH * HID + (long)(b0 + mr) * HID + nn] = t;
      }
    }
    __syncthreads();
    p ^= 1;
#pragma unroll
    for (int r = 0; r < 8; ++r)
      xp[r] = xpn[r];
  }
}

// ---------------------------------------------------------------------------
extern "C" void kernel_launch(void* const* d_in, const int* in_sizes, int n_in,
                              void* d_out, int out_size, void* d_ws, size_t ws_size,
                              hipStream_t stream) {
  (void)in_sizes; (void)n_in; (void)out_size; (void)d_ws; (void)ws_size;
  const float* x  = (const float*)d_in[0];
  const float* Wx = (const float*)d_in[1];
  const float* bx = (const float*)d_in[2];
  const float* Wh = (const float*)d_in[3];
  const float* bh = (const float*)d_in[4];
  float* out = (float*)d_out;

  // Phase 1: xp into d_out's h_seq region (memory-bound, massively parallel).
  rnn_proj_kernel<<<dim3((SEQ * BATCH) / 16), dim3(256), 0, stream>>>(
      x, Wx, bx, bh, out);

  // Phase 2: serial scan, batch-parallel across 16 workgroups.
  rnn_scan_kernel<<<dim3(BATCH / 16), dim3(256), 0, stream>>>(Wh, out);
}
